// CausalSelfAttention_79156247266046
// MI455X (gfx1250) — compile-verified, matrix operand order checked by reading, not run on hardware
//
#include <hip/hip_runtime.h>

// ---------------------------------------------------------------------------
// Types
// ---------------------------------------------------------------------------
typedef __attribute__((ext_vector_type(16))) __bf16          v16bf;
typedef __attribute__((ext_vector_type(8)))  float           v8f;
typedef __attribute__((ext_vector_type(16))) unsigned short  v16u;
typedef __attribute__((ext_vector_type(8)))  unsigned short  v8u;

static __device__ __forceinline__ v16bf as_bf(v16u u) {
    union { v16u u; v16bf b; } c; c.u = u; return c.b;
}

static __device__ __forceinline__ v8f wmma_bf16(v16u a, v16u b, v8f c) {
    // D = A(16x32 bf16) * B(32x16 bf16) + C(16x16 f32)
    return __builtin_amdgcn_wmma_f32_16x16x32_bf16(
        /*neg_a=*/false, as_bf(a), /*neg_b=*/false, as_bf(b),
        /*c_mod=*/(short)0, c, /*reuse_a=*/false, /*reuse_b=*/false);
}

static __device__ __forceinline__ unsigned short f2bf(float f) {
    unsigned int u = __float_as_uint(f);
    unsigned int r = u + 0x7FFFu + ((u >> 16) & 1u);  // round-to-nearest-even
    return (unsigned short)(r >> 16);
}

// Load a 16-bit WMMA operand fragment: per lane two contiguous 16B chunks
// (elements 0..7 = K[hi*8..hi*8+7], elements 8..15 = K[16+hi*8..]).
static __device__ __forceinline__ v16u ldfrag(const unsigned short* p0,
                                              const unsigned short* p1) {
    v8u lo = *(const v8u*)p0;
    v8u hi = *(const v8u*)p1;
    return __builtin_shufflevector(lo, hi, 0,1,2,3,4,5,6,7,8,9,10,11,12,13,14,15);
}

static __device__ __forceinline__ float rmax16(float v) {
    v = fmaxf(v, __shfl_xor(v, 1, 32));
    v = fmaxf(v, __shfl_xor(v, 2, 32));
    v = fmaxf(v, __shfl_xor(v, 4, 32));
    v = fmaxf(v, __shfl_xor(v, 8, 32));
    return v;
}
static __device__ __forceinline__ float rsum16(float v) {
    v += __shfl_xor(v, 1, 32);
    v += __shfl_xor(v, 2, 32);
    v += __shfl_xor(v, 4, 32);
    v += __shfl_xor(v, 8, 32);
    return v;
}

// gfx1250 async DMA: copy 16B per lane from global memory into LDS, tracked by
// ASYNCcnt (CDNA5 ISA). lds_off is a DS-space byte
// offset (low 32 bits of a generic __shared__ pointer), gaddr a 64-bit global
// address.
static __device__ __forceinline__ void async_cp16(unsigned lds_off,
                                                  const void* gaddr) {
    asm volatile("global_load_async_to_lds_b128 %0, %1, off"
                 :: "v"(lds_off), "v"((unsigned long long)(size_t)gaddr)
                 : "memory");
}

// ---------------------------------------------------------------------------
// fp32 -> bf16 conversion
// ---------------------------------------------------------------------------
__global__ __launch_bounds__(256) void cvt_bf16_kernel(const float* __restrict__ in,
                                                       unsigned short* __restrict__ out,
                                                       int n) {
    int i = blockIdx.x * 256 + threadIdx.x;
    if (i < n) out[i] = f2bf(in[i]);
}

// ---------------------------------------------------------------------------
// GEMM: C[M,N] (f32) = A[M,K] (bf16 row-major) * B[N,K]^T (bf16 row-major)
// Block = 256 threads = 8 waves; block tile 128x128; wave tile 64x32 (4x2 WMMA).
// A/B panels (128x32) are staged into double-buffered LDS via
// global_load_async_to_lds_b128 so the DMA of chunk k+1 overlaps the WMMAs of
// chunk k. LDS row pitch = 40 ushorts (20 DWORDs) -> the 16 row-parallel
// ds_load_b128 fragment reads hit distinct banks.
// ---------------------------------------------------------------------------
#define LPITCH 40

__global__ __launch_bounds__(256) void gemm_bf16_kernel(
    const unsigned short* __restrict__ A,
    const unsigned short* __restrict__ B,
    float* __restrict__ C,
    int M, int N, int K) {
    __shared__ __align__(16) unsigned short sA[2][128][LPITCH];
    __shared__ __align__(16) unsigned short sB[2][128][LPITCH];

    const int t    = threadIdx.x;
    const int lane = t & 31;
    const int wv   = t >> 5;
    const int wm   = wv >> 2;           // 0..1
    const int wn   = wv & 3;            // 0..3
    const int ml   = lane & 15;
    const int hi   = lane >> 4;

    const int m0b = blockIdx.y * 128;
    const int n0b = blockIdx.x * 128;

    const unsigned aBase = (unsigned)(size_t)&sA[0][0][0];
    const unsigned bBase = (unsigned)(size_t)&sB[0][0][0];
    const unsigned bufStride = 128u * LPITCH * 2u;   // bytes per buffer

    // Copy map: 128 rows x 32 cols bf16 = 512 x 16B chunks per matrix;
    // thread t moves chunks u = t and u = 256+t (row = u>>2, 16B-col = u&3).
    const int r0 = t >> 2,          c0 = t & 3;
    const int r1 = (256 + t) >> 2,  c1 = (256 + t) & 3;

    v8f acc[4][2];
    const v8f vzero = {0.f,0.f,0.f,0.f,0.f,0.f,0.f,0.f};
    #pragma unroll
    for (int mf = 0; mf < 4; ++mf)
        #pragma unroll
        for (int nf = 0; nf < 2; ++nf) acc[mf][nf] = vzero;

    const int nkc = K >> 5;     // 32-wide K chunks

    // Prologue: DMA chunk 0 into buffer 0.
    {
        const int k0 = 0;
        async_cp16(aBase + (unsigned)(r0 * LPITCH + c0 * 8) * 2,
                   A + (size_t)(m0b + r0) * K + k0 + c0 * 8);
        async_cp16(bBase + (unsigned)(r0 * LPITCH + c0 * 8) * 2,
                   B + (size_t)(n0b + r0) * K + k0 + c0 * 8);
        async_cp16(aBase + (unsigned)(r1 * LPITCH + c1 * 8) * 2,
                   A + (size_t)(m0b + r1) * K + k0 + c1 * 8);
        async_cp16(bBase + (unsigned)(r1 * LPITCH + c1 * 8) * 2,
                   B + (size_t)(n0b + r1) * K + k0 + c1 * 8);
    }

    for (int kc = 0; kc < nkc; ++kc) {
        const int cur = kc & 1;
        const bool has_next = (kc + 1) < nkc;
        if (has_next) {
            const int nxt = cur ^ 1;
            const int k0 = (kc + 1) << 5;
            const unsigned ao = aBase + (unsigned)nxt * bufStride;
            const unsigned bo = bBase + (unsigned)nxt * bufStride;
            async_cp16(ao + (unsigned)(r0 * LPITCH + c0 * 8) * 2,
                       A + (size_t)(m0b + r0) * K + k0 + c0 * 8);
            async_cp16(bo + (unsigned)(r0 * LPITCH + c0 * 8) * 2,
                       B + (size_t)(n0b + r0) * K + k0 + c0 * 8);
            async_cp16(ao + (unsigned)(r1 * LPITCH + c1 * 8) * 2,
                       A + (size_t)(m0b + r1) * K + k0 + c1 * 8);
            async_cp16(bo + (unsigned)(r1 * LPITCH + c1 * 8) * 2,
                       B + (size_t)(n0b + r1) * K + k0 + c1 * 8);
            // 4 newer copies outstanding; <=4 means chunk kc's copies landed
            // (async loads complete in order).
            asm volatile("s_wait_asynccnt 0x4" ::: "memory");
        } else {
            asm volatile("s_wait_asynccnt 0x0" ::: "memory");
        }
        __syncthreads();

        // Fragments from LDS + 8 WMMAs for this 32-wide K chunk.
        v16u a[4], b[2];
        #pragma unroll
        for (int mf = 0; mf < 4; ++mf) {
            const unsigned short* pr = &sA[cur][wm * 64 + mf * 16 + ml][0];
            a[mf] = ldfrag(pr + hi * 8, pr + 16 + hi * 8);
        }
        #pragma unroll
        for (int nf = 0; nf < 2; ++nf) {
            const unsigned short* pr = &sB[cur][wn * 32 + nf * 16 + ml][0];
            b[nf] = ldfrag(pr + hi * 8, pr + 16 + hi * 8);
        }
        #pragma unroll
        for (int mf = 0; mf < 4; ++mf)
            #pragma unroll
            for (int nf = 0; nf < 2; ++nf)
                acc[mf][nf] = wmma_bf16(a[mf], b[nf], acc[mf][nf]);

        __syncthreads();    // before buffer `cur` is DMA-overwritten at kc+2
    }

    // C/D layout: lane l holds col (l&15); VGPR r -> row (l>>4)*8 + r.
    #pragma unroll
    for (int mf = 0; mf < 4; ++mf) {
        #pragma unroll
        for (int nf = 0; nf < 2; ++nf) {
            #pragma unroll
            for (int r = 0; r < 8; ++r) {
                int m = m0b + wm * 64 + mf * 16 + hi * 8 + r;
                int n = n0b + wn * 32 + nf * 16 + ml;
                C[(size_t)m * N + n] = acc[mf][nf][r];
            }
        }
    }
}

// ---------------------------------------------------------------------------
// RoPE + layout: qkv f32 [B][S][3072] ->
//   qb  bf16 [B][32][S][64]   (rope)
//   kb  bf16 [B][8][S][64]    (rope)
//   vtb bf16 [B][8][64][S]    (transposed V for P*V B-operand)
// ---------------------------------------------------------------------------
__global__ __launch_bounds__(256) void rope_layout_kernel(
    const float* __restrict__ qkv,
    const float* __restrict__ cosb,
    const float* __restrict__ sinb,
    unsigned short* __restrict__ qb,
    unsigned short* __restrict__ kb,
    unsigned short* __restrict__ vtb) {
    int idx = blockIdx.x * 256 + threadIdx.x;   // B*32*2048*64 threads
    int d = idx & 63;
    int s = (idx >> 6) & 2047;
    int h = (idx >> 17) & 31;
    int b = idx >> 22;

    float cs = cosb[s * 64 + d];
    float sn = sinb[s * 64 + d];
    const float* row = qkv + ((size_t)(b * 2048 + s)) * 3072;

    float tq = row[h * 64 + d];
    float rq = (d < 32) ? -row[h * 64 + d + 32] : row[h * 64 + d - 32];
    qb[((size_t)((b * 32 + h) * 2048 + s)) * 64 + d] = f2bf(tq * cs + rq * sn);

    if (h < 8) {
        const float* krow = row + 2048 + h * 64;
        float tk = krow[d];
        float rk = (d < 32) ? -krow[d + 32] : krow[d - 32];
        kb[((size_t)((b * 8 + h) * 2048 + s)) * 64 + d] = f2bf(tk * cs + rk * sn);

        float vv = row[2560 + h * 64 + d];
        vtb[((size_t)((b * 8 + h) * 64 + d)) * 2048 + s] = f2bf(vv);
    }
}

// ---------------------------------------------------------------------------
// Flash attention: one wave per (b, h, 16-row q-tile). Streams 32-wide k-tiles
// up to the causal boundary. Output: ob bf16 [B][S][H*64].
// ---------------------------------------------------------------------------
__global__ __launch_bounds__(256) void attn_kernel(
    const unsigned short* __restrict__ qb,
    const unsigned short* __restrict__ kb,
    const unsigned short* __restrict__ vtb,
    unsigned short* __restrict__ ob) {
    __shared__ __align__(16) unsigned short sp[8][16][32];  // per-wave P staging

    const int lane = threadIdx.x & 31;
    const int wv   = threadIdx.x >> 5;
    const int ml   = lane & 15;
    const int hi   = lane >> 4;

    const int w  = blockIdx.x * 8 + wv;         // 0 .. 8191
    const int qt = w & 127;                     // q tile (16 rows)
    const int h  = (w >> 7) & 31;
    const int b  = w >> 12;
    const int kvh = h >> 2;                     // GQA: n_rep = 4

    const unsigned short* qrow = qb + ((size_t)((b * 32 + h) * 2048 + qt * 16 + ml)) * 64;
    const v16u aq0 = ldfrag(qrow + hi * 8,      qrow + 16 + hi * 8);   // d = 0..31
    const v16u aq1 = ldfrag(qrow + 32 + hi * 8, qrow + 48 + hi * 8);   // d = 32..63

    const unsigned short* kbase = kb  + ((size_t)((b * 8 + kvh) * 2048)) * 64;
    const unsigned short* vbase = vtb + ((size_t)((b * 8 + kvh) * 64))  * 2048;

    const v8f vzero = {0.f,0.f,0.f,0.f,0.f,0.f,0.f,0.f};
    v8f o[4];
    #pragma unroll
    for (int dt = 0; dt < 4; ++dt) o[dt] = vzero;
    float mrow[8], lrow[8];
    #pragma unroll
    for (int r = 0; r < 8; ++r) { mrow[r] = -3.0e38f; lrow[r] = 0.0f; }

    const int nkt = (qt * 16 + 47) >> 5;        // number of 32-wide k-tiles

    for (int kt = 0; kt < nkt; ++kt) {
        // ---- scores S = Q * K^T  (two 16x16 tiles, K-dim = 64 via 2 chunks)
        const unsigned short* kr0 = kbase + (size_t)(kt * 32 + ml) * 64;
        const unsigned short* kr1 = kbase + (size_t)(kt * 32 + 16 + ml) * 64;
        v16u bk00 = ldfrag(kr0 + hi * 8,      kr0 + 16 + hi * 8);
        v16u bk01 = ldfrag(kr0 + 32 + hi * 8, kr0 + 48 + hi * 8);
        v16u bk10 = ldfrag(kr1 + hi * 8,      kr1 + 16 + hi * 8);
        v16u bk11 = ldfrag(kr1 + 32 + hi * 8, kr1 + 48 + hi * 8);

        v8f s0 = vzero, s1 = vzero;
        s0 = wmma_bf16(aq0, bk00, s0);
        s0 = wmma_bf16(aq1, bk01, s0);
        s1 = wmma_bf16(aq0, bk10, s1);
        s1 = wmma_bf16(aq1, bk11, s1);

        // ---- online softmax over this 16x32 tile
        const int rbase = qt * 16 + hi * 8;
        const int c0 = kt * 32 + ml;
        const int c1 = c0 + 16;
        #pragma unroll
        for (int r = 0; r < 8; ++r) {
            const int rg = rbase + r;
            float v0 = s0[r] * 0.125f; if (c0 > rg) v0 = -1.0e30f;
            float v1 = s1[r] * 0.125f; if (c1 > rg) v1 = -1.0e30f;
            float mnew = fmaxf(mrow[r], rmax16(fmaxf(v0, v1)));
            float corr = __expf(mrow[r] - mnew);
            float e0 = __expf(v0 - mnew);
            float e1 = __expf(v1 - mnew);
            lrow[r] = lrow[r] * corr + rsum16(e0 + e1);
            mrow[r] = mnew;
            o[0][r] *= corr; o[1][r] *= corr; o[2][r] *= corr; o[3][r] *= corr;
            // C-layout -> LDS [row][col] (bf16) for A-layout reload
            sp[wv][hi * 8 + r][ml]      = f2bf(e0);
            sp[wv][hi * 8 + r][16 + ml] = f2bf(e1);
        }
        asm volatile("s_wait_dscnt 0x0" ::: "memory");  // wave-internal LDS RAW

        // ---- P (16x32 bf16, A-layout) from LDS
        const unsigned short* pr = &sp[wv][ml][0];
        v16u ap = ldfrag(pr + hi * 8, pr + 16 + hi * 8);

        // ---- O += P * V  (V transposed: row-major [d][s] == [N][K])
        #pragma unroll
        for (int dt = 0; dt < 4; ++dt) {
            const unsigned short* vr = vbase + (size_t)(dt * 16 + ml) * 2048 + kt * 32;
            v16u bv = ldfrag(vr + hi * 8, vr + 16 + hi * 8);
            o[dt] = wmma_bf16(ap, bv, o[dt]);
        }
    }

    // ---- epilogue: O /= l, store bf16 to [B][S][H*64]
    unsigned short* obase = ob + ((size_t)(b * 2048 + qt * 16)) * 2048 + h * 64;
    #pragma unroll
    for (int r = 0; r < 8; ++r) {
        float inv = 1.0f / lrow[r];
        unsigned short* orow = obase + (size_t)(hi * 8 + r) * 2048;
        #pragma unroll
        for (int dt = 0; dt < 4; ++dt)
            orow[dt * 16 + ml] = f2bf(o[dt][r] * inv);
    }
}

// ---------------------------------------------------------------------------
// Launch
// ---------------------------------------------------------------------------
extern "C" void kernel_launch(void* const* d_in, const int* in_sizes, int n_in,
                              void* d_out, int out_size, void* d_ws, size_t ws_size,
                              hipStream_t stream) {
    const float* x     = (const float*)d_in[0];   // [2][2048][2048]
    const float* cosb  = (const float*)d_in[1];   // [2048][64]
    const float* sinb  = (const float*)d_in[2];   // [2048][64]
    const float* w_qkv = (const float*)d_in[3];   // [3072][2048]
    const float* w_out = (const float*)d_in[4];   // [2048][2048]
    float* out = (float*)d_out;                   // [2][2048][2048]

    char* ws = (char*)d_ws;
    // workspace layout (bytes)
    unsigned short* xb    = (unsigned short*)(ws);                 // 16,777,216
    unsigned short* wqkvb = (unsigned short*)(ws +  16777216);     // 12,582,912
    unsigned short* woutb = (unsigned short*)(ws +  29360128);     //  8,388,608
    float*          qkvf  = (float*)         (ws +  37748736);     // 50,331,648
    unsigned short* qbp   = (unsigned short*)(ws +  88080384);     // 16,777,216
    unsigned short* kbp   = (unsigned short*)(ws + 104857600);     //  4,194,304
    unsigned short* vtp   = (unsigned short*)(ws + 109051904);     //  4,194,304
    unsigned short* atp   = (unsigned short*)(ws + 113246208);     // 16,777,216

    // 1) fp32 -> bf16
    cvt_bf16_kernel<<<32768, 256, 0, stream>>>(x,     xb,    8388608);
    cvt_bf16_kernel<<<24576, 256, 0, stream>>>(w_qkv, wqkvb, 6291456);
    cvt_bf16_kernel<<<16384, 256, 0, stream>>>(w_out, woutb, 4194304);

    // 2) QKV projection: [4096,3072] = xb[4096,2048] * wqkvb[3072,2048]^T
    gemm_bf16_kernel<<<dim3(24, 32), 256, 0, stream>>>(xb, wqkvb, qkvf,
                                                       4096, 3072, 2048);

    // 3) RoPE + q/k/v layouts (V transposed)
    rope_layout_kernel<<<32768, 256, 0, stream>>>(qkvf, cosb, sinb, qbp, kbp, vtp);

    // 4) causal flash attention (8192 waves, 8 per block)
    attn_kernel<<<1024, 256, 0, stream>>>(qbp, kbp, vtp, atp);

    // 5) output projection: [4096,2048] = atp[4096,2048] * woutb[2048,2048]^T
    gemm_bf16_kernel<<<dim3(16, 32), 256, 0, stream>>>(atp, woutb, out,
                                                       4096, 2048, 2048);
}
